// Disc_edge2_15573551415685
// MI455X (gfx1250) — matrix operand order repaired and checked by reading, not compiled, not run on hardware
//
#include <hip/hip_runtime.h>

// ---------------------------------------------------------------------------
// GNN2 (dense batched graphs) for MI455X / gfx1250, wave32 + WMMA bf16.
// B=16, N=128, DN0=64, DE0=16, DH=128.
// msg = relu([e | x_j] @ [We;Wj] + xi + b) * A   (xj folded into the WMMA)
// ---------------------------------------------------------------------------

#define B_  16
#define N_  128
#define DH_ 128

typedef __attribute__((ext_vector_type(16))) __bf16 v16bf;
typedef __attribute__((ext_vector_type(8)))  float  v8f;

union Frag {
    v16bf        v;
    uint4        q[2];
    unsigned int u[8];
};

__device__ __forceinline__ unsigned short f2bf(float f) {
    unsigned int u = __float_as_uint(f);
    unsigned int r = (u + 0x7fffu + ((u >> 16) & 1u)) >> 16;  // RNE
    return (unsigned short)r;
}
__device__ __forceinline__ float bf2f(unsigned short s) {
    return __uint_as_float(((unsigned int)s) << 16);
}
__device__ __forceinline__ unsigned int pack2(float lo, float hi) {
    return (unsigned int)f2bf(lo) | ((unsigned int)f2bf(hi) << 16);
}

// ---------------------------------------------------------------------------
// Generic fp32 -> bf16 conversion (for x tensors).
// ---------------------------------------------------------------------------
__global__ __launch_bounds__(256) void cvt_bf16_kernel(
    const float* __restrict__ in, unsigned short* __restrict__ out, int n)
{
    const int i = blockIdx.x * 256 + threadIdx.x;
    if (i < n) out[i] = f2bf(in[i]);
}

// ---------------------------------------------------------------------------
// Build combined B-matrix [We ; Wj ; 0-pad] as bf16 [128][Kpad], k contiguous.
// We = wmsg[2*Dn : 2*Dn+De], Wj = wmsg[Dn : 2*Dn].
// ---------------------------------------------------------------------------
__global__ __launch_bounds__(256) void convert_w_kernel(
    const float* __restrict__ wmsg, int Dn, int De, int total,
    unsigned short* __restrict__ out)
{
    const int idx = blockIdx.x * 256 + threadIdx.x;   // over Kpad*DH
    if (idx >= total) return;
    const int k = idx >> 7;
    const int h = idx & (DH_ - 1);
    float v = 0.0f;
    if (k < De)           v = wmsg[(size_t)(2 * Dn + k) * DH_ + h];       // We
    else if (k < De + Dn) v = wmsg[(size_t)(Dn + (k - De)) * DH_ + h];    // Wj
    out[(size_t)h * (total >> 7) + k] = f2bf(v);
}

// ---------------------------------------------------------------------------
// Edge kernel. One workgroup per (b,i); 8 waves, wave w owns j-tile [16w,16w+16).
// A-operand row j = [ e[b,i,j,:] | xbf[b,j,:] | pad ].
// MODE 0: layer 0  — e fp32 De=16, Dn=64, K=96;  writes msg bf16 + agg.
// MODE 1: layer 1  — e bf16 K=256; writes emix=0.5*(e+msg) bf16 IN PLACE + agg.
// MODE 2: layer 2  — e bf16 K=256; no msg write; accumulates hsum[b,h].
// Output strip is staged per-wave in LDS, flushed with coalesced b128 stores.
// ---------------------------------------------------------------------------
template <int MODE>
__global__ __launch_bounds__(256) void edge_kernel(
    const int*            __restrict__ A,        // [B,N,N]
    const float*          __restrict__ e32,      // MODE0 input
    const unsigned short* __restrict__ ebf,      // MODE1/2 input
    const unsigned short* __restrict__ xbf,      // [B,N,Dn] bf16
    const float*          __restrict__ xi,       // [B,N,DH]
    const unsigned short* __restrict__ wcomb,    // [128][Kpad] bf16
    const float*          __restrict__ bmsg,     // [DH]
    unsigned short*       __restrict__ msg_out,  // MODE0/1 output (MODE1: ==ebf)
    float*                __restrict__ agg_out,  // [B,N,DH]   (MODE0/1)
    float*                __restrict__ hsum)     // [B,DH]     (MODE2)
{
    constexpr int  KSTEPS  = (MODE == 0) ? 3 : 8;
    constexpr int  EPITCH  = (MODE == 0) ? 104 : 264;  // bf16 elems; %64B==16B
    constexpr bool WRITES  = (MODE != 2);
    constexpr int  STPITCH = 136;                      // out-strip pitch (bf16)

    __shared__ unsigned short lds_e[N_ * EPITCH];
    __shared__ unsigned short lds_st[WRITES ? (8 * 16 * STPITCH) : 1];
    __shared__ float lds_agg[DH_];
    __shared__ float lds_xi[DH_];
    __shared__ float lds_bias[DH_];

    const int wg   = blockIdx.x;       // b*N + i
    const int b    = wg >> 7;
    const int tid  = threadIdx.x;
    const int lane = tid & 31;
    const int wave = tid >> 5;
    const int half = (lane >> 4) & 1;
    const int ln   = lane & 15;

    if (tid < DH_) {
        lds_xi[tid]   = xi[(size_t)wg * DH_ + tid];
        lds_bias[tid] = bmsg[tid];
        lds_agg[tid]  = 0.0f;
    }

    // ---- stage A-operand rows [e | x | pad] into LDS as bf16 ----
    {
        const int j   = tid >> 1;
        const int sub = tid & 1;
        if (MODE == 0) {
            // e: 16 fp32 -> bf16 at [0..15]
            if (sub == 0) {
                const float4* src = (const float4*)(e32 + (((size_t)wg * N_) + j) * 16);
                float4 p0 = src[0], p1 = src[1], p2 = src[2], p3 = src[3];
                unsigned int* dst = (unsigned int*)&lds_e[j * EPITCH];
                dst[0] = pack2(p0.x, p0.y); dst[1] = pack2(p0.z, p0.w);
                dst[2] = pack2(p1.x, p1.y); dst[3] = pack2(p1.z, p1.w);
                dst[4] = pack2(p2.x, p2.y); dst[5] = pack2(p2.z, p2.w);
                dst[6] = pack2(p3.x, p3.y); dst[7] = pack2(p3.z, p3.w);
            } else {
                // zero pad k = 80..95
                uint4* dz = (uint4*)&lds_e[j * EPITCH + 80];
                dz[0] = make_uint4(0u,0u,0u,0u);
                dz[1] = make_uint4(0u,0u,0u,0u);
            }
            // x: 64 bf16 at [16..79]; each sub copies 32 bf16
            const uint4* xs = (const uint4*)(xbf + (((size_t)b * N_) + j) * 64 + sub * 32);
            uint4* xd = (uint4*)&lds_e[j * EPITCH + 16 + sub * 32];
            xd[0] = xs[0]; xd[1] = xs[1];
        } else {
            const int koff = sub * 64;
            // e: 128 bf16 at [0..127]
            const uint4* es = (const uint4*)(ebf + (((size_t)wg * N_) + j) * DH_ + koff);
            uint4* ed = (uint4*)&lds_e[j * EPITCH + koff];
            ed[0] = es[0]; ed[1] = es[1]; ed[2] = es[2]; ed[3] = es[3];
            // x: 128 bf16 at [128..255]
            const uint4* xs = (const uint4*)(xbf + (((size_t)b * N_) + j) * DH_ + koff);
            uint4* xd = (uint4*)&lds_e[j * EPITCH + 128 + koff];
            xd[0] = xs[0]; xd[1] = xs[1]; xd[2] = xs[2]; xd[3] = xs[3];
            // zero pad k = 256..263
            if (sub) *(uint4*)&lds_e[j * EPITCH + 256] = make_uint4(0u,0u,0u,0u);
        }
    }
    __syncthreads();

    // ---- per-wave adjacency mask (rows j0+8*half+r) ----
    const int j0 = wave * 16;
    const int* Arow = A + (size_t)wg * N_;
    float am[8];
    #pragma unroll
    for (int r = 0; r < 8; ++r) am[r] = (float)Arow[j0 + half * 8 + r];

    // ---- GEMM: ks outer, 8 h-tile accumulators live; A frag loaded once/ks ----
    v8f c[8];
    {
        v8f z = {};
        #pragma unroll
        for (int t = 0; t < 8; ++t) c[t] = z;
    }
    #pragma unroll
    for (int ks = 0; ks < KSTEPS; ++ks) {
        Frag a;
        const int k0 = ks * 32 + half * 8;
        a.q[0] = *(const uint4*)&lds_e[(j0 + ln) * EPITCH + k0];
        a.q[1] = *(const uint4*)&lds_e[(j0 + ln) * EPITCH + k0 + 16];

        const unsigned short* wrow = wcomb + ks * 32 + half * 8;
        #pragma unroll
        for (int ht = 0; ht < 8; ++ht) {
            Frag bf;
            bf.q[0] = *(const uint4*)&wrow[(size_t)(ht * 16 + ln) * (KSTEPS * 32)];
            bf.q[1] = *(const uint4*)&wrow[(size_t)(ht * 16 + ln) * (KSTEPS * 32) + 16];
            c[ht] = __builtin_amdgcn_wmma_f32_16x16x32_bf16(
                        false, a.v, false, bf.v, (short)0, c[ht], false, false);
        }
    }

    // ---- epilogue: + xi + bias, relu, mask; agg; stage out-strip in LDS ----
    unsigned short* strow = WRITES ? &lds_st[wave * 16 * STPITCH] : (unsigned short*)nullptr;
    #pragma unroll
    for (int ht = 0; ht < 8; ++ht) {
        const int   h  = ht * 16 + ln;            // C/D: lane holds column h0+(lane&15)
        const float ab = lds_xi[h] + lds_bias[h];
        float accum = 0.0f;
        #pragma unroll
        for (int r = 0; r < 8; ++r) {             // C/D: VGPR r -> row r + 8*half
            const int jr = half * 8 + r;          // row within wave's 16-row strip
            float v = fmaxf(c[ht][r] + ab, 0.0f) * am[r];
            accum += v;
            if (MODE == 0) {
                strow[jr * STPITCH + h] = f2bf(v);
            } else if (MODE == 1) {
                const float prev = bf2f(lds_e[(j0 + jr) * EPITCH + h]);
                strow[jr * STPITCH + h] = f2bf(0.5f * (prev + v));
            }
        }
        atomicAdd(&lds_agg[h], accum);
    }

    // ---- flush wave's 16x128 strip with coalesced b128 stores ----
    if (WRITES) {
        unsigned short* mrow = msg_out + (size_t)wg * N_ * DH_;
        const int row = lane >> 1;
        const int seg = lane & 1;
        const uint4* s = (const uint4*)&strow[row * STPITCH + seg * 64];
        uint4 q0 = s[0], q1 = s[1], q2 = s[2], q3 = s[3];
        uint4* g = (uint4*)(mrow + (size_t)(j0 + row) * DH_ + seg * 64);
        g[0] = q0; g[1] = q1; g[2] = q2; g[3] = q3;
    }
    __syncthreads();

    if (tid < DH_) {
        if (MODE == 2) atomicAdd(&hsum[b * DH_ + tid], lds_agg[tid]);
        else           agg_out[(size_t)wg * DH_ + tid] = lds_agg[tid];
    }
}

// ---------------------------------------------------------------------------
// xi = x @ Wi  (Wi = w_msg[:Dn])
// ---------------------------------------------------------------------------
__global__ __launch_bounds__(256) void xi_kernel(
    const float* __restrict__ x, int Dn,
    const float* __restrict__ wmsg,
    float* __restrict__ xi)
{
    const int idx = blockIdx.x * 256 + threadIdx.x;   // over B*N*DH
    const int h   = idx & (DH_ - 1);
    const int row = idx >> 7;
    const float* xr = x + (size_t)row * Dn;
    float acc = 0.0f;
    for (int k = 0; k < Dn; ++k) acc = fmaf(xr[k], wmsg[(size_t)k * DH_ + h], acc);
    xi[idx] = acc;
}

// ---------------------------------------------------------------------------
// x_new = relu(x@Wx + agg@Wa + b); optional output mix 0.5*(mix_with + x_new)
// ---------------------------------------------------------------------------
__global__ __launch_bounds__(256) void node_kernel(
    const float* __restrict__ x, int Dn,
    const float* __restrict__ agg,
    const float* __restrict__ wnode, const float* __restrict__ bnode,
    const float* __restrict__ mix_with,
    float* __restrict__ out)
{
    const int idx = blockIdx.x * 256 + threadIdx.x;
    const int h   = idx & (DH_ - 1);
    const int row = idx >> 7;
    const float* xr = x + (size_t)row * Dn;
    const float* ar = agg + (size_t)row * DH_;
    float acc = bnode[h];
    for (int k = 0; k < Dn; ++k)  acc = fmaf(xr[k], wnode[(size_t)k * DH_ + h], acc);
    const float* Wa = wnode + (size_t)Dn * DH_;
    for (int k = 0; k < DH_; ++k) acc = fmaf(ar[k], Wa[(size_t)k * DH_ + h], acc);
    acc = fmaxf(acc, 0.0f);
    if (mix_with) acc = 0.5f * (mix_with[idx] + acc);
    out[idx] = acc;
}

__global__ void zero_kernel(float* p, int n) {
    int i = blockIdx.x * 256 + threadIdx.x;
    if (i < n) p[i] = 0.0f;
}

// ---------------------------------------------------------------------------
// Head: h = hsum/(N*N); relu MLP x2; out[b] = h@w_h3 + b_h3. One block per b.
// ---------------------------------------------------------------------------
__global__ __launch_bounds__(128) void head_kernel(
    const float* __restrict__ hsum,
    const float* __restrict__ w1, const float* __restrict__ b1,
    const float* __restrict__ w2, const float* __restrict__ b2,
    const float* __restrict__ w3, const float* __restrict__ b3,
    float* __restrict__ out)
{
    __shared__ float ha[DH_], hb[DH_], red[128];
    const int b = blockIdx.x, t = threadIdx.x;
    ha[t] = hsum[b * DH_ + t] * (1.0f / (float)(N_ * N_));
    __syncthreads();
    float acc = b1[t];
    for (int k = 0; k < DH_; ++k) acc = fmaf(ha[k], w1[k * DH_ + t], acc);
    hb[t] = fmaxf(acc, 0.0f);
    __syncthreads();
    acc = b2[t];
    for (int k = 0; k < DH_; ++k) acc = fmaf(hb[k], w2[k * DH_ + t], acc);
    red[t] = fmaxf(acc, 0.0f) * w3[t];
    __syncthreads();
    for (int s = 64; s > 0; s >>= 1) {
        if (t < s) red[t] += red[t + s];
        __syncthreads();
    }
    if (t == 0) out[b] = red[0] + b3[0];
}

// ---------------------------------------------------------------------------
extern "C" void kernel_launch(void* const* d_in, const int* in_sizes, int n_in,
                              void* d_out, int out_size, void* d_ws, size_t ws_size,
                              hipStream_t stream)
{
    (void)in_sizes; (void)n_in; (void)out_size; (void)ws_size;

    const int*   A        = (const int*)  d_in[0];
    const float* x0       = (const float*)d_in[1];
    const float* e0       = (const float*)d_in[2];
    const float* w_msg_0  = (const float*)d_in[3];
    const float* b_msg_0  = (const float*)d_in[4];
    const float* w_node_0 = (const float*)d_in[5];
    const float* b_node_0 = (const float*)d_in[6];
    const float* w_msg_1  = (const float*)d_in[7];
    const float* b_msg_1  = (const float*)d_in[8];
    const float* w_node_1 = (const float*)d_in[9];
    const float* b_node_1 = (const float*)d_in[10];
    const float* w_msg_2  = (const float*)d_in[11];
    const float* b_msg_2  = (const float*)d_in[12];
    const float* w_h1     = (const float*)d_in[15];
    const float* b_h1     = (const float*)d_in[16];
    const float* w_h2     = (const float*)d_in[17];
    const float* b_h2     = (const float*)d_in[18];
    const float* w_h3     = (const float*)d_in[19];
    const float* b_h3     = (const float*)d_in[20];
    float* out = (float*)d_out;

    // workspace layout
    char* ws = (char*)d_ws;
    const size_t MSG_BYTES = (size_t)B_ * N_ * N_ * DH_ * sizeof(unsigned short); // 64 MiB
    const size_t VEC_BYTES = (size_t)B_ * N_ * DH_ * sizeof(float);               // 1 MiB
    unsigned short* msgbuf = (unsigned short*)ws;                 ws += MSG_BYTES;
    float* xi   = (float*)ws;  ws += VEC_BYTES;
    float* agg  = (float*)ws;  ws += VEC_BYTES;
    float* x1   = (float*)ws;  ws += VEC_BYTES;
    float* xmix = (float*)ws;  ws += VEC_BYTES;
    float* hsum = (float*)ws;  ws += 256 * sizeof(float) * 8;     // 16*128 floats
    unsigned short* wcomb = (unsigned short*)ws;                  // up to 256x128 bf16
    ws += (size_t)256 * DH_ * sizeof(unsigned short);             // 64 KiB
    unsigned short* xbf = (unsigned short*)ws;                    // [B,N,<=128] bf16
    ws += (size_t)B_ * N_ * DH_ * sizeof(unsigned short);         // 512 KiB

    const dim3 blk256(256), blk128(128);
    const dim3 grid_vec(B_ * N_ * DH_ / 256);
    const dim3 grid_edge(B_ * N_);

    // ---- layer 0 (Dn=64, De=16, Kpad=96) ----
    cvt_bf16_kernel<<<B_ * N_ * 64 / 256, blk256, 0, stream>>>(x0, xbf, B_ * N_ * 64);
    xi_kernel<<<grid_vec, blk256, 0, stream>>>(x0, 64, w_msg_0, xi);
    convert_w_kernel<<<48, blk256, 0, stream>>>(w_msg_0, 64, 16, 96 * DH_, wcomb);
    edge_kernel<0><<<grid_edge, blk256, 0, stream>>>(
        A, e0, nullptr, xbf, xi, wcomb, b_msg_0, msgbuf, agg, nullptr);
    node_kernel<<<grid_vec, blk256, 0, stream>>>(
        x0, 64, agg, w_node_0, b_node_0, nullptr, x1);

    // ---- layer 1 (writes emix = 0.5*(msg0+msg1) in place into msgbuf) ----
    cvt_bf16_kernel<<<grid_vec, blk256, 0, stream>>>(x1, xbf, B_ * N_ * DH_);
    xi_kernel<<<grid_vec, blk256, 0, stream>>>(x1, 128, w_msg_1, xi);
    convert_w_kernel<<<128, blk256, 0, stream>>>(w_msg_1, 128, 128, 256 * DH_, wcomb);
    edge_kernel<1><<<grid_edge, blk256, 0, stream>>>(
        A, nullptr, msgbuf, xbf, xi, wcomb, b_msg_1, msgbuf, agg, nullptr);
    node_kernel<<<grid_vec, blk256, 0, stream>>>(
        x1, 128, agg, w_node_1, b_node_1, x1, xmix);

    // ---- layer 2 (no msg/agg materialization; hsum only) ----
    cvt_bf16_kernel<<<grid_vec, blk256, 0, stream>>>(xmix, xbf, B_ * N_ * DH_);
    xi_kernel<<<grid_vec, blk256, 0, stream>>>(xmix, 128, w_msg_2, xi);
    convert_w_kernel<<<128, blk256, 0, stream>>>(w_msg_2, 128, 128, 256 * DH_, wcomb);
    zero_kernel<<<8, blk256, 0, stream>>>(hsum, B_ * DH_);
    edge_kernel<2><<<grid_edge, blk256, 0, stream>>>(
        A, nullptr, msgbuf, xbf, xi, wcomb, b_msg_2, nullptr, nullptr, hsum);

    // ---- head MLP ----
    head_kernel<<<B_, blk128, 0, stream>>>(
        hsum, w_h1, b_h1, w_h2, b_h2, w_h3, b_h3, out);
}
